// GraphAttentionEncoder_16312285791053
// MI455X (gfx1250) — matrix operand order repaired and checked by reading, not compiled, not run on hardware
//
#include <hip/hip_runtime.h>
#include <hip/hip_bf16.h>

// ---------------------------------------------------------------------------
// GraphAttentionEncoder for MI455X (gfx1250).
//
// Algorithmic transform: LN is row-wise, so LN(x[col]) @ Wn == (LN(x)@Wn)[col].
// Per-NODE projections + per-edge gather cuts matrix FLOPs ~12x (168 -> 13.5
// GFLOP).  Everything is L2-resident (192 MB L2).
//
// GEMMs: v_wmma_f32_16x16x32_bf16.  Activations are stored once as bf16 by
// their producers (no per-GEMM conversion ALU in the hot loop).  B panels are
// staged into double-buffered LDS per block (8x less L2 traffic, latency
// hidden behind WMMAs), using the gfx1250 async global->LDS path
// (global_load_async_to_lds_b128, ASYNCcnt-tracked) when declared.
// ---------------------------------------------------------------------------

#define N_NODES 10000
#define KNBR    16
#define DFEAT   512
#define ZOUT    64
#define RADIUS_ 10.0f
#define ALPHA_  2.0f
#define THR_    0.01f

typedef __attribute__((ext_vector_type(16))) __bf16 v16bf;
typedef __attribute__((ext_vector_type(8)))  float  v8f;
typedef unsigned short u16;

union BfVec { unsigned u[8]; v16bf v; };

// Detect the gfx1250 async global->LDS builtins (ASYNCcnt-tracked DMA).
#if defined(__has_builtin)
#  if __has_builtin(__builtin_amdgcn_global_load_async_to_lds_b128)
#    define HAVE_ASYNC_COPY 1
#  endif
#endif
#ifndef HAVE_ASYNC_COPY
#  define HAVE_ASYNC_COPY 0
#endif

#if HAVE_ASYNC_COPY
// Builtin prototype is "vV4i*1V4i*3IiIi": (global int4* src, local int4* dst,
// imm offset, imm cpol).  V4i is a GCC-style (generic) vector in clang's
// builtin type decoder, so use vector_size, not ext_vector_type.
typedef int v4i_ __attribute__((vector_size(16)));
typedef __attribute__((address_space(1))) v4i_ Gv4i;
typedef __attribute__((address_space(3))) v4i_ Lv4i;
__device__ __forceinline__ void async_b128_to_lds(const void* gsrc, void* ldst) {
  // generic->AS casts via integer round trip (LDS generic addr low 32 bits
  // are the LDS byte offset on gfx12 apertures).
  __builtin_amdgcn_global_load_async_to_lds_b128(
      (Gv4i*)(uintptr_t)gsrc,
      (Lv4i*)(uintptr_t)(unsigned)(uintptr_t)ldst, 0, 0);
}
__device__ __forceinline__ void wait_async0() {
#  if __has_builtin(__builtin_amdgcn_s_wait_asynccnt)
  __builtin_amdgcn_s_wait_asynccnt(0);
#  else
  asm volatile("s_wait_asynccnt 0x0" ::: "memory");
#  endif
}
#endif

// f32 -> bf16 round-to-nearest-even (bit-level)
__device__ __forceinline__ unsigned f2bf(float f) {
  unsigned u = __float_as_uint(f);
  return (u + 0x7FFFu + ((u >> 16) & 1u)) >> 16;
}

// ---------------------------------------------------------------------------
// Pack row-major f32 W[K=DFEAT][ncols] into WMMA bf16 B-operand layout:
// tile (nt,kt); lane L: n = nt*16+(L&15), h=L>>4 -> K range [kt*32+16h,+16),
// VGPR v = K pair (2v,2v+1) packed in one u32.  8 u32/lane, contiguous.
// ---------------------------------------------------------------------------
__global__ void pack_w_kernel(const float* __restrict__ W,
                              unsigned* __restrict__ P, int ncols) {
  int w = blockIdx.x * blockDim.x + threadIdx.x;
  int words = (DFEAT * ncols) >> 1;
  if (w >= words) return;
  int v      = w & 7;
  int lane   = (w >> 3) & 31;
  int tile   = w >> 8;                 // = nt*ktiles + kt
  int ktiles = DFEAT / 32;             // 16
  int kt     = tile % ktiles;
  int nt     = tile / ktiles;
  int n = nt * 16 + (lane & 15);
  int h = lane >> 4;
  int k = kt * 32 + 16 * h + 2 * v;
  unsigned lo = f2bf(W[(size_t)k * ncols + n]);
  unsigned hi = f2bf(W[(size_t)(k + 1) * ncols + n]);
  P[w] = lo | (hi << 16);
}

// ---------------------------------------------------------------------------
// LayerNorm: one wave per row, __shfl_xor tree reduction; writes f32 (for the
// edge phase) and bf16 (A operand for the WMMA GEMMs).
// ---------------------------------------------------------------------------
__global__ void ln_kernel(const float* __restrict__ X, const float* __restrict__ g,
                          const float* __restrict__ b, float* __restrict__ T,
                          u16* __restrict__ Tb, int nrows) {
  int wave = threadIdx.x >> 5;
  int lane = threadIdx.x & 31;
  int row  = blockIdx.x * 8 + wave;
  if (row >= nrows) return;
  const float* xr = X + (size_t)row * DFEAT;
  float v[16];
  float s1 = 0.f, s2 = 0.f;
#pragma unroll
  for (int j = 0; j < 16; ++j) {
    float x = xr[lane + 32 * j];
    v[j] = x; s1 += x; s2 += x * x;
  }
#pragma unroll
  for (int m = 16; m >= 1; m >>= 1) {
    s1 += __shfl_xor(s1, m, 32);
    s2 += __shfl_xor(s2, m, 32);
  }
  float mean = s1 * (1.0f / DFEAT);
  float var  = s2 * (1.0f / DFEAT) - mean * mean;
  float inv  = rsqrtf(var + 1e-5f);
  float* tr = T + (size_t)row * DFEAT;
  u16*  trb = Tb + (size_t)row * DFEAT;
#pragma unroll
  for (int j = 0; j < 16; ++j) {
    int c = lane + 32 * j;
    float o = (v[j] - mean) * inv * g[c] + b[c];
    tr[c]  = o;
    trb[c] = (u16)f2bf(o);
  }
}

// ---------------------------------------------------------------------------
// WMMA GEMM: C[M x ncols] = act(A_bf16[M x 512] * W + bias).
// Block = 256 thr = 8 waves; wave owns a 16(M) x 64(N) strip.  Per k-step the
// block stages the 4 B tiles (4 KB) into double-buffered LDS (async DMA when
// available), issued BEFORE the WMMAs of the current step so the copy hides
// behind matrix work.  A fragment = two contiguous b128 loads of bf16.
// ACT: 0 none, 1 leaky(leaky(z)) = z>=0?z:1e-4z, 2 leaky.
// ---------------------------------------------------------------------------
template <int ACT>
__global__ void __launch_bounds__(256)
gemm_kernel(const u16* __restrict__ A, const unsigned* __restrict__ Bp,
            const float* __restrict__ bias, float* __restrict__ Cf,
            u16* __restrict__ Cb, int M, int ncols) {
  __shared__ __align__(16) unsigned sB[2][1024];  // 2 x 4 KB

  const int ktiles = DFEAT / 32;  // 16
  int tid  = threadIdx.x;
  int wave = tid >> 5;
  int lane = tid & 31;
  int h    = lane >> 4;
  int ln   = lane & 15;
  int n0   = blockIdx.y * 64;

  int strip   = blockIdx.x * 8 + wave;
  bool active = (strip * 16 < M);
  int m0      = active ? strip * 16 : 0;   // inactive waves compute row 0,
                                           // skip stores; all hit barriers.
  const u16* arow = A + (size_t)(m0 + ln) * DFEAT;

  // Staging map: 4 tiles x 64 threads; each thread moves one b128.
  int s_tile = tid >> 6;
  int s_thr  = tid & 63;
  int s_lane = s_thr >> 1;
  int s_half = s_thr & 1;
  const unsigned* gsrc =
      Bp + ((size_t)((n0 >> 4) + s_tile) * ktiles * 32 + s_lane) * 8 +
      s_half * 4;                                   // + kt*256 per k-step
  unsigned lw = s_tile * 256 + s_lane * 8 + s_half * 4;

  // ---- prologue: stage kt = 0 into buffer 0 ----
#if HAVE_ASYNC_COPY
  async_b128_to_lds(gsrc, &sB[0][lw]);
  wait_async0();
#else
  {
    uint4 st = *(const uint4*)gsrc;
    *(uint4*)&sB[0][lw] = st;
  }
#endif
  __syncthreads();

  v8f acc[4] = {};
  for (int kt = 0; kt < ktiles; ++kt) {
    int buf = kt & 1;
    // ---- stage next k-step into the other buffer (overlaps WMMAs) ----
#if HAVE_ASYNC_COPY
    if (kt + 1 < ktiles)
      async_b128_to_lds(gsrc + (size_t)(kt + 1) * 256, &sB[buf ^ 1][lw]);
#else
    uint4 stn;
    if (kt + 1 < ktiles) stn = *(const uint4*)(gsrc + (size_t)(kt + 1) * 256);
#endif
    __builtin_prefetch(arow + (kt + 1) * 32, 0, 0);  // global_prefetch_b8

    // ---- A fragment: two b128 loads of pre-converted bf16 ----
    uint4 alo = *(const uint4*)(arow + kt * 32 + 16 * h);
    uint4 ahi = *(const uint4*)(arow + kt * 32 + 16 * h + 8);
    BfVec av;
    av.u[0] = alo.x; av.u[1] = alo.y; av.u[2] = alo.z; av.u[3] = alo.w;
    av.u[4] = ahi.x; av.u[5] = ahi.y; av.u[6] = ahi.z; av.u[7] = ahi.w;

#pragma unroll
    for (int t = 0; t < 4; ++t) {
      const uint4* bp = (const uint4*)&sB[buf][t * 256 + lane * 8];
      uint4 b0 = bp[0], b1 = bp[1];
      BfVec bv;
      bv.u[0] = b0.x; bv.u[1] = b0.y; bv.u[2] = b0.z; bv.u[3] = b0.w;
      bv.u[4] = b1.x; bv.u[5] = b1.y; bv.u[6] = b1.z; bv.u[7] = b1.w;
      acc[t] = __builtin_amdgcn_wmma_f32_16x16x32_bf16(
          false, av.v, false, bv.v, (short)0, acc[t], false, false);
    }

#if HAVE_ASYNC_COPY
    if (kt + 1 < ktiles) wait_async0();
#else
    if (kt + 1 < ktiles) *(uint4*)&sB[buf ^ 1][lw] = stn;
#endif
    __syncthreads();
  }

  if (!active) return;
  // Epilogue.  C/D layout: VGPR v, lane L -> row m0+8*(L>>4)+v,
  // col n0+16*t+(L&15).
#pragma unroll
  for (int t = 0; t < 4; ++t) {
    int n = n0 + 16 * t + ln;
    float bb = bias[n];
#pragma unroll
    for (int v = 0; v < 8; ++v) {
      int m = m0 + 8 * h + v;
      float z = acc[t][v] + bb;
      if (ACT == 1) z = (z >= 0.f) ? z : z * 1.0e-4f;  // leaky(leaky(z))
      else if (ACT == 2) z = (z >= 0.f) ? z : z * 0.01f;
      Cf[(size_t)m * ncols + n] = z;
      if (Cb) Cb[(size_t)m * ncols + n] = (u16)f2bf(z);
    }
  }
}

// ---------------------------------------------------------------------------
// Edge phase: block per node; gathers NS[col] / Tn[col] (L2-resident rows),
// 17-way per-channel softmax with thresholding; writes ctx as bf16 (it is
// only ever consumed as a WMMA A operand).
// ---------------------------------------------------------------------------
__global__ void edge_kernel(const float* __restrict__ S,
                            const float* __restrict__ NS,
                            const float* __restrict__ Tt,
                            const float* __restrict__ Tn,
                            const float* __restrict__ coords,
                            const int* __restrict__ col,
                            const float* __restrict__ beta_p,
                            u16* __restrict__ ctxb) {
  __shared__ int   s_col[KNBR];
  __shared__ float s_dw[KNBR];
  int i   = blockIdx.x;
  int tid = threadIdx.x;
  if (tid < KNBR) {
    int j = col[(size_t)i * KNBR + tid];
    s_col[tid] = j;
    float dx = coords[2 * i]     - coords[2 * j];
    float dy = coords[2 * i + 1] - coords[2 * j + 1];
    float dist = sqrtf(dx * dx + dy * dy);
    s_dw[tid] = __expf(-ALPHA_ * dist / (RADIUS_ + 1e-8f));
  }
  __syncthreads();
  float beta = beta_p[0];
  float ob   = 1.0f - beta;

#pragma unroll
  for (int half = 0; half < 2; ++half) {
    int c = tid + half * 256;
    float e[KNBR + 1];
    e[0] = S[(size_t)i * DFEAT + c];
    float mx = e[0];
#pragma unroll
    for (int k = 0; k < KNBR; ++k) {
      float v = NS[(size_t)s_col[k] * DFEAT + c] * s_dw[k];
      e[k + 1] = v;
      mx = fmaxf(mx, v);
    }
    float sum = 0.f;
#pragma unroll
    for (int k = 0; k <= KNBR; ++k) {
      e[k] = __expf(e[k] - mx);
      sum += e[k];
    }
    float inv = 1.0f / sum;
    float sw = e[0] * inv;
    if (sw < THR_) sw = 0.f;
    float acc = beta * sw * Tt[(size_t)i * DFEAT + c];
#pragma unroll
    for (int k = 0; k < KNBR; ++k) {
      float nw = e[k + 1] * inv;
      if (nw < THR_) nw = 0.f;
      acc += ob * nw * Tn[(size_t)s_col[k] * DFEAT + c];
    }
    ctxb[(size_t)i * DFEAT + c] = (u16)f2bf(acc);
  }
}

// ---------------------------------------------------------------------------
extern "C" void kernel_launch(void* const* d_in, const int* in_sizes, int n_in,
                              void* d_out, int out_size, void* d_ws,
                              size_t ws_size, hipStream_t stream) {
  const float* x      = (const float*)d_in[0];
  const float* coords = (const float*)d_in[1];
  const int*   edge   = (const int*)d_in[2];
  const float* g1  = (const float*)d_in[3];
  const float* b1  = (const float*)d_in[4];
  const float* Ws1 = (const float*)d_in[5];
  const float* bs1 = (const float*)d_in[6];
  const float* Wn1 = (const float*)d_in[7];
  const float* bn1 = (const float*)d_in[8];
  const float* Wr1 = (const float*)d_in[9];
  const float* br1 = (const float*)d_in[10];
  const float* be1 = (const float*)d_in[11];
  const float* g2  = (const float*)d_in[12];
  const float* b2  = (const float*)d_in[13];
  const float* Ws2 = (const float*)d_in[14];
  const float* bs2 = (const float*)d_in[15];
  const float* Wn2 = (const float*)d_in[16];
  const float* bn2 = (const float*)d_in[17];
  const float* Wr2 = (const float*)d_in[18];
  const float* br2 = (const float*)d_in[19];
  const float* be2 = (const float*)d_in[20];
  const float* Wo  = (const float*)d_in[21];
  const float* bo  = (const float*)d_in[22];

  // Workspace: packed weights (~3.2 MB) + 5 f32 buffers + 4 bf16 buffers.
  char* ws = (char*)d_ws;
  const size_t PW = (size_t)DFEAT * DFEAT * 2;  // 512 KB packed bf16 weight
  unsigned* pWs1 = (unsigned*)(ws);
  unsigned* pWn1 = (unsigned*)(ws + PW);
  unsigned* pWr1 = (unsigned*)(ws + 2 * PW);
  unsigned* pWs2 = (unsigned*)(ws + 3 * PW);
  unsigned* pWn2 = (unsigned*)(ws + 4 * PW);
  unsigned* pWr2 = (unsigned*)(ws + 5 * PW);
  unsigned* pWo  = (unsigned*)(ws + 6 * PW);
  const size_t FB = (size_t)N_NODES * DFEAT * 4;  // 20.48 MB (f32)
  const size_t HB = (size_t)N_NODES * DFEAT * 2;  // 10.24 MB (bf16)
  char* fb = ws + 4 * 1024 * 1024;
  float* T1f  = (float*)(fb);            // T1, reused as T2t
  float* Sf   = (float*)(fb + FB);       // S1/S2
  float* NSf  = (float*)(fb + 2 * FB);   // NS1/NS2
  float* H1f  = (float*)(fb + 3 * FB);   // H1, reused as H2 f32 scratch
  float* T2nf = (float*)(fb + 4 * FB);
  char* hb = fb + 5 * FB;
  u16* T1b  = (u16*)(hb);                // T1 bf16, reused as T2t bf16
  u16* CTXb = (u16*)(hb + HB);
  u16* T2nb = (u16*)(hb + 2 * HB);
  u16* H2b  = (u16*)(hb + 3 * HB);

  const int* col = edge + (size_t)N_NODES * KNBR;  // second row of edge_index

  dim3 blk(256);
  int pbDD = (DFEAT * DFEAT / 2 + 255) / 256;
  int pbO  = (DFEAT * ZOUT / 2 + 255) / 256;
  pack_w_kernel<<<pbDD, blk, 0, stream>>>(Ws1, pWs1, DFEAT);
  pack_w_kernel<<<pbDD, blk, 0, stream>>>(Wn1, pWn1, DFEAT);
  pack_w_kernel<<<pbDD, blk, 0, stream>>>(Wr1, pWr1, DFEAT);
  pack_w_kernel<<<pbDD, blk, 0, stream>>>(Ws2, pWs2, DFEAT);
  pack_w_kernel<<<pbDD, blk, 0, stream>>>(Wn2, pWn2, DFEAT);
  pack_w_kernel<<<pbDD, blk, 0, stream>>>(Wr2, pWr2, DFEAT);
  pack_w_kernel<<<pbO,  blk, 0, stream>>>(Wo,  pWo,  ZOUT);

  dim3 lnGrid((N_NODES + 7) / 8);
  dim3 gGrid((N_NODES / 16 + 7) / 8, DFEAT / 64);
  dim3 oGrid((N_NODES / 16 + 7) / 8, ZOUT / 64);

  // --- block 1 ---
  ln_kernel<<<lnGrid, blk, 0, stream>>>(x, g1, b1, T1f, T1b, N_NODES);
  gemm_kernel<0><<<gGrid, blk, 0, stream>>>(T1b, pWs1, bs1, Sf,  nullptr, N_NODES, DFEAT);
  gemm_kernel<0><<<gGrid, blk, 0, stream>>>(T1b, pWn1, bn1, NSf, nullptr, N_NODES, DFEAT);
  edge_kernel<<<N_NODES, blk, 0, stream>>>(Sf, NSf, T1f, T1f, coords, col, be1, CTXb);
  gemm_kernel<1><<<gGrid, blk, 0, stream>>>(CTXb, pWr1, br1, H1f, nullptr, N_NODES, DFEAT);

  // --- block 2 (targets = H1, neighbors = original x) ---
  ln_kernel<<<lnGrid, blk, 0, stream>>>(H1f, g2, b2, T1f,  T1b,  N_NODES);  // T2t
  ln_kernel<<<lnGrid, blk, 0, stream>>>(x,   g2, b2, T2nf, T2nb, N_NODES);
  gemm_kernel<0><<<gGrid, blk, 0, stream>>>(T1b,  pWs2, bs2, Sf,  nullptr, N_NODES, DFEAT);
  gemm_kernel<0><<<gGrid, blk, 0, stream>>>(T2nb, pWn2, bn2, NSf, nullptr, N_NODES, DFEAT);
  edge_kernel<<<N_NODES, blk, 0, stream>>>(Sf, NSf, T1f, T2nf, coords, col, be2, CTXb);
  gemm_kernel<1><<<gGrid, blk, 0, stream>>>(CTXb, pWr2, br2, H1f, H2b, N_NODES, DFEAT);

  // --- output projection ---
  gemm_kernel<2><<<oGrid, blk, 0, stream>>>(H2b, pWo, bo, (float*)d_out,
                                            nullptr, N_NODES, ZOUT);
}
